// SAGEResBNNet_71313636983150
// MI455X (gfx1250) — matrix operand order repaired and checked by reading, not compiled, not run on hardware
//
#include <hip/hip_runtime.h>
#include <stdint.h>
#include <stddef.h>

#define NND 50000
#define NE  800000
#define DIN 166
#define DINP 192
#define HID 128
#define NCLS 2
#define NCLSP 16
#define BN_EPS 1e-5f
#define LDSPAD 200   // ushorts per LDS row: 192 data + 8 pad (400B -> conflict-free 16-row frag reads)

typedef __attribute__((ext_vector_type(16))) __bf16 v16bf;
typedef __attribute__((ext_vector_type(8)))  float  v8f;

union FragB { v16bf v; uint4 q[2]; };

__device__ __forceinline__ unsigned short f32_to_bf16_rne(float f) {
  unsigned u = __float_as_uint(f);
  u += 0x7FFFu + ((u >> 16) & 1u);
  return (unsigned short)(u >> 16);
}

// ---------------- utility kernels ----------------

__global__ void zero_u32(unsigned* p, long long n) {
  long long i  = (long long)blockIdx.x * blockDim.x + threadIdx.x;
  long long st = (long long)gridDim.x * blockDim.x;
  for (; i < n; i += st) p[i] = 0u;
}

__global__ void count_deg(const int* __restrict__ dst, int* __restrict__ cnt) {
  int e = blockIdx.x * blockDim.x + threadIdx.x;
  if (e < NE) atomicAdd(&cnt[dst[e]], 1);
}

// convert float [rows x cols] -> bf16 [rowspad x colspad], zero padded
__global__ void cvt_pad_bf16(const float* __restrict__ in, unsigned short* __restrict__ out,
                             int rows, int cols, int rowspad, int colspad) {
  int tid = blockIdx.x * blockDim.x + threadIdx.x;
  int total = rowspad * colspad;
  if (tid >= total) return;
  int r = tid / colspad, c = tid - r * colspad;
  float v = (r < rows && c < cols) ? in[(size_t)r * cols + c] : 0.f;
  out[tid] = f32_to_bf16_rne(v);
}

// generic edge-parallel scatter add: one wave per edge, lanes stride feature dim
__global__ void scatter_add(const float* __restrict__ h, const int* __restrict__ src,
                            const int* __restrict__ dst, float* __restrict__ acc, int d) {
  int tid = blockIdx.x * blockDim.x + threadIdx.x;
  int e = __builtin_amdgcn_readfirstlane(tid >> 5);   // wave-uniform -> SGPR
  int lane = tid & 31;
  if (e >= NE) return;
  int s = src[e], t = dst[e];
  const float* hp = h + (size_t)s * d;
  float* ap = acc + (size_t)t * d;
  for (int f = lane; f < d; f += 32) atomicAdd(ap + f, hp[f]);
}

// specialized d=128: lane handles 4 contiguous floats (b128 load + 4 atomics)
__global__ void scatter_add128(const float* __restrict__ h, const int* __restrict__ src,
                               const int* __restrict__ dst, float* __restrict__ acc) {
  int tid = blockIdx.x * blockDim.x + threadIdx.x;
  int e = __builtin_amdgcn_readfirstlane(tid >> 5);   // wave-uniform -> SGPR
  int lane = tid & 31;
  if (e >= NE) return;
  int s = src[e], t = dst[e];
  const float4 v = *(const float4*)(h + (size_t)s * HID + lane * 4);
  float* ap = acc + (size_t)t * HID + lane * 4;
  atomicAdd(ap + 0, v.x);
  atomicAdd(ap + 1, v.y);
  atomicAdd(ap + 2, v.z);
  atomicAdd(ap + 3, v.w);
}

// acc/max(cnt,1) -> bf16 padded [NND x dpad]
__global__ void finalize_mean(const float* __restrict__ acc, const int* __restrict__ cnt,
                              unsigned short* __restrict__ outb, int d, int dpad) {
  int tid = blockIdx.x * blockDim.x + threadIdx.x;
  if (tid >= NND * dpad) return;
  int r = tid / dpad, c = tid - r * dpad;
  float v = 0.f;
  if (c < d) {
    float cn = (float)cnt[r];
    if (cn < 1.f) cn = 1.f;
    v = acc[(size_t)r * d + c] / cn;
  }
  outb[tid] = f32_to_bf16_rne(v);
}

// ---------------- WMMA GEMM with LDS-staged A tiles ----------------
// Block = 256 threads = 8 waves; block handles one M tile (16 rows), wave w
// handles N tile w (w < ntiles_n). A1/A2 tiles staged once into LDS and shared
// by all 8 waves; W streamed from global (small, L2-resident).
// 50000 = 3125*16 exact -> EXEC all-1s in every wave issuing WMMA.
__global__ void __launch_bounds__(256)
wmma_gemm(const unsigned short* __restrict__ A1, int K1, const unsigned short* __restrict__ W1,
          const unsigned short* __restrict__ A2, int K2, const unsigned short* __restrict__ W2,
          const float* __restrict__ bias,
          const float* __restrict__ bn_g, const float* __restrict__ bn_b,
          const float* __restrict__ bn_m, const float* __restrict__ bn_v,
          const float* __restrict__ res,
          float* __restrict__ outf, unsigned short* __restrict__ outb,
          int ntiles_n, int dout_store, int out_stride) {
  __shared__ unsigned short lA[2][16 * LDSPAD];   // 2 x 6.25 KB

  const int tm  = blockIdx.x;
  const int tid = threadIdx.x;

  // cooperative staging: A tiles -> LDS (uint4 = 8 ushorts per copy)
  {
    const int kv = K1 >> 3;            // vec4 groups per row
    const int nvec = 16 * kv;
    for (int i = tid; i < nvec; i += 256) {
      int r = i / kv, c = (i - r * kv) << 3;
      *(uint4*)&lA[0][r * LDSPAD + c] =
          *(const uint4*)(A1 + (size_t)(tm * 16 + r) * K1 + c);
    }
  }
  if (A2 != nullptr) {
    const int kv = K2 >> 3;
    const int nvec = 16 * kv;
    for (int i = tid; i < nvec; i += 256) {
      int r = i / kv, c = (i - r * kv) << 3;
      *(uint4*)&lA[1][r * LDSPAD + c] =
          *(const uint4*)(A2 + (size_t)(tm * 16 + r) * K2 + c);
    }
  }
  __syncthreads();

  const int wave = tid >> 5;
  const int lane = tid & 31;
  if (wave >= ntiles_n) return;        // wave-uniform exit (layer 2: 1 tile)

  const int tn   = wave;
  const int half = lane >> 4;          // 0: lanes 0-15, 1: lanes 16-31
  const int r15  = lane & 15;
  const int kA = half << 3;            // A frag K base: 0 / 8
  const int kB = half << 4;            // unused; B frag also uses kA (0 / 8)

  v8f acc = {0.f, 0.f, 0.f, 0.f, 0.f, 0.f, 0.f, 0.f};

  {
    const unsigned short* Ar = &lA[0][r15 * LDSPAD];
    const unsigned short* Wr = W1 + (size_t)(tn * 16 + r15) * K1;
    for (int kk = 0; kk < K1; kk += 32) {
      FragB fa, fb;
      fa.q[0] = *(const uint4*)(Ar + kk + kA);        // K = base..base+7
      fa.q[1] = *(const uint4*)(Ar + kk + kA + 16);   // K = base+16..base+23
      fb.q[0] = *(const uint4*)(Wr + kk + kA);        // K = base..base+7
      fb.q[1] = *(const uint4*)(Wr + kk + kA + 16);   // K = base+16..base+23
      acc = __builtin_amdgcn_wmma_f32_16x16x32_bf16(false, fa.v, false, fb.v,
                                                    (short)0, acc, false, false);
    }
  }
  if (A2 != nullptr) {
    const unsigned short* Ar = &lA[1][r15 * LDSPAD];
    const unsigned short* Wr = W2 + (size_t)(tn * 16 + r15) * K2;
    for (int kk = 0; kk < K2; kk += 32) {
      FragB fa, fb;
      fa.q[0] = *(const uint4*)(Ar + kk + kA);
      fa.q[1] = *(const uint4*)(Ar + kk + kA + 16);
      fb.q[0] = *(const uint4*)(Wr + kk + kA);
      fb.q[1] = *(const uint4*)(Wr + kk + kA + 16);
      acc = __builtin_amdgcn_wmma_f32_16x16x32_bf16(false, fa.v, false, fb.v,
                                                    (short)0, acc, false, false);
    }
  }

  const int col = tn * 16 + r15;       // D layout: N = lane&15
  const bool has_bn = (bn_g != nullptr);
  float scale = 1.f, shift = 0.f;
  if (has_bn) {
    float g = bn_g[col];
    scale = g * rsqrtf(bn_v[col] + BN_EPS);
    shift = bn_b[col] - bn_m[col] * scale;
  }
  float bv = 0.f;
  if (bias != nullptr && col < dout_store) bv = bias[col];

  if (col < dout_store) {
#pragma unroll
    for (int i = 0; i < 8; i++) {
      int row = tm * 16 + i + (half << 3);  // D layout: M = i + 8*half
      float v = acc[i] + bv;
      if (has_bn) { v = v * scale + shift; v = v > 0.f ? v : 0.f; }
      if (res != nullptr) v += res[(size_t)row * out_stride + col];
      outf[(size_t)row * out_stride + col] = v;
      if (outb != nullptr) outb[(size_t)row * out_stride + col] = f32_to_bf16_rne(v);
    }
  }
}

// ---------------- host launcher ----------------

extern "C" void kernel_launch(void* const* d_in, const int* in_sizes, int n_in,
                              void* d_out, int out_size, void* d_ws, size_t ws_size,
                              hipStream_t stream) {
  const float* x    = (const float*)d_in[0];
  const int*   ei   = (const int*)d_in[1];
  const int*   src  = ei;
  const int*   dst  = ei + NE;
  const float* Wl0  = (const float*)d_in[2];
  const float* bl0  = (const float*)d_in[3];
  const float* Wr0  = (const float*)d_in[4];
  const float* Wl1  = (const float*)d_in[5];
  const float* bl1  = (const float*)d_in[6];
  const float* Wr1  = (const float*)d_in[7];
  const float* Wl2  = (const float*)d_in[8];
  const float* bl2  = (const float*)d_in[9];
  const float* Wr2  = (const float*)d_in[10];
  const float* Wres0 = (const float*)d_in[11];
  const float* bn0g = (const float*)d_in[12];
  const float* bn0b = (const float*)d_in[13];
  const float* bn0m = (const float*)d_in[14];
  const float* bn0v = (const float*)d_in[15];
  const float* bn1g = (const float*)d_in[16];
  const float* bn1b = (const float*)d_in[17];
  const float* bn1m = (const float*)d_in[18];
  const float* bn1v = (const float*)d_in[19];

  // ---- workspace carve (aligned 256B) ----
  char* p = (char*)d_ws;
  auto carve = [&](size_t bytes) -> char* {
    char* r = p;
    p += (bytes + 255) & ~(size_t)255;
    return r;
  };
  int*            cnt   = (int*)carve((size_t)NND * 4);
  float*          acc   = (float*)carve((size_t)NND * DIN * 4);        // reused for 128-wide layers
  unsigned short* xb    = (unsigned short*)carve((size_t)NND * DINP * 2);
  unsigned short* aggb  = (unsigned short*)carve((size_t)NND * DINP * 2);
  float*          h0f   = (float*)carve((size_t)NND * HID * 4);
  unsigned short* h0b   = (unsigned short*)carve((size_t)NND * HID * 2);
  float*          h1f   = (float*)carve((size_t)NND * HID * 4);        // also resbuf for layer 0
  unsigned short* h1b   = (unsigned short*)carve((size_t)NND * HID * 2);
  unsigned short* Wl0b  = (unsigned short*)carve((size_t)HID * DINP * 2);
  unsigned short* Wr0b  = (unsigned short*)carve((size_t)HID * DINP * 2);
  unsigned short* Wrsb  = (unsigned short*)carve((size_t)HID * DINP * 2);
  unsigned short* Wl1b  = (unsigned short*)carve((size_t)HID * HID * 2);
  unsigned short* Wr1b  = (unsigned short*)carve((size_t)HID * HID * 2);
  unsigned short* Wl2b  = (unsigned short*)carve((size_t)NCLSP * HID * 2);
  unsigned short* Wr2b  = (unsigned short*)carve((size_t)NCLSP * HID * 2);

  const int B = 256;
  auto g1 = [](long long n, int b) { return (unsigned)((n + b - 1) / b); };

  // degree counts (shared by all layers)
  zero_u32<<<g1(NND, B) > 4096u ? 4096u : g1(NND, B), B, 0, stream>>>((unsigned*)cnt, NND);
  count_deg<<<g1(NE, B), B, 0, stream>>>(dst, cnt);

  // bf16 conversions of x and all weights
  cvt_pad_bf16<<<g1((long long)NND * DINP, B), B, 0, stream>>>(x, xb, NND, DIN, NND, DINP);
  cvt_pad_bf16<<<g1((long long)HID * DINP, B), B, 0, stream>>>(Wl0, Wl0b, HID, DIN, HID, DINP);
  cvt_pad_bf16<<<g1((long long)HID * DINP, B), B, 0, stream>>>(Wr0, Wr0b, HID, DIN, HID, DINP);
  cvt_pad_bf16<<<g1((long long)HID * DINP, B), B, 0, stream>>>(Wres0, Wrsb, HID, DIN, HID, DINP);
  cvt_pad_bf16<<<g1((long long)HID * HID, B), B, 0, stream>>>(Wl1, Wl1b, HID, HID, HID, HID);
  cvt_pad_bf16<<<g1((long long)HID * HID, B), B, 0, stream>>>(Wr1, Wr1b, HID, HID, HID, HID);
  cvt_pad_bf16<<<g1((long long)NCLSP * HID, B), B, 0, stream>>>(Wl2, Wl2b, NCLS, HID, NCLSP, HID);
  cvt_pad_bf16<<<g1((long long)NCLSP * HID, B), B, 0, stream>>>(Wr2, Wr2b, NCLS, HID, NCLSP, HID);

  dim3 blk(256, 1, 1);
  dim3 gemm_grid(NND / 16, 1, 1);           // 3125 blocks, 8 waves = 8 N tiles

  // ---------- layer 0 ----------
  zero_u32<<<4096u, B, 0, stream>>>((unsigned*)acc, (long long)NND * DIN);
  scatter_add<<<g1((long long)NE * 32, B), B, 0, stream>>>(x, src, dst, acc, DIN);
  finalize_mean<<<g1((long long)NND * DINP, B), B, 0, stream>>>(acc, cnt, aggb, DIN, DINP);

  // residual projection: res0 = x @ Wres0^T  (into h1f, free until layer 1)
  wmma_gemm<<<gemm_grid, blk, 0, stream>>>(
      xb, DINP, Wrsb, nullptr, 0, nullptr,
      nullptr, nullptr, nullptr, nullptr, nullptr, nullptr,
      h1f, nullptr, HID / 16, HID, HID);

  // h0 = relu(bn0(agg@Wl0^T + x@Wr0^T + bl0)) + res0
  wmma_gemm<<<gemm_grid, blk, 0, stream>>>(
      aggb, DINP, Wl0b, xb, DINP, Wr0b,
      bl0, bn0g, bn0b, bn0m, bn0v, h1f,
      h0f, h0b, HID / 16, HID, HID);

  // ---------- layer 1 ----------
  zero_u32<<<4096u, B, 0, stream>>>((unsigned*)acc, (long long)NND * HID);
  scatter_add128<<<g1((long long)NE * 32, B), B, 0, stream>>>(h0f, src, dst, acc);
  finalize_mean<<<g1((long long)NND * HID, B), B, 0, stream>>>(acc, cnt, aggb, HID, HID);

  // h1 = relu(bn1(agg@Wl1^T + h0@Wr1^T + bl1)) + h0
  wmma_gemm<<<gemm_grid, blk, 0, stream>>>(
      aggb, HID, Wl1b, h0b, HID, Wr1b,
      bl1, bn1g, bn1b, bn1m, bn1v, h0f,
      h1f, h1b, HID / 16, HID, HID);

  // ---------- layer 2 (output) ----------
  zero_u32<<<4096u, B, 0, stream>>>((unsigned*)acc, (long long)NND * HID);
  scatter_add128<<<g1((long long)NE * 32, B), B, 0, stream>>>(h1f, src, dst, acc);
  finalize_mean<<<g1((long long)NND * HID, B), B, 0, stream>>>(acc, cnt, aggb, HID, HID);

  // out = agg@Wl2^T + h1@Wr2^T + bl2   (store 2 of 16 cols)
  wmma_gemm<<<gemm_grid, blk, 0, stream>>>(
      aggb, HID, Wl2b, h1b, HID, Wr2b,
      bl2, nullptr, nullptr, nullptr, nullptr, nullptr,
      (float*)d_out, nullptr, 1, NCLS, NCLS);
}
